// GINJK_30305289241327
// MI455X (gfx1250) — matrix operand order repaired and checked by reading, not compiled, not run on hardware
//
#include <hip/hip_runtime.h>
#include <cstddef>

typedef __bf16 bf16;
typedef __attribute__((ext_vector_type(8)))  bf16  bf16x8;
typedef __attribute__((ext_vector_type(16))) bf16  bf16x16;
typedef __attribute__((ext_vector_type(8)))  float f32x8;

constexpr int kNodes   = 100000;
constexpr int kEdges   = 1600000;
constexpr int kHid     = 128;
constexpr int kHid2    = 256;
constexpr int kLayers  = 4;
constexpr int kClasses = 10;
constexpr int kGraphs  = 512;
constexpr int kCat     = kHid * kLayers;   // 512
constexpr float kBnEps = 1e-5f;

// ---------------------------------------------------------------- helpers
__device__ __forceinline__ unsigned short f2bf(float f) {
  unsigned int u = __float_as_uint(f);
  unsigned int r = u + 0x7FFFu + ((u >> 16) & 1u);   // round-to-nearest-even
  return (unsigned short)(r >> 16);
}

__global__ void zerof_kernel(float* __restrict__ p, long n) {
  long i = (long)blockIdx.x * blockDim.x + threadIdx.x;
  if (i < n) p[i] = 0.0f;
}

// ---------------------------------------------------------------- degree / counts
__global__ void deg_kernel(const int* __restrict__ dst, float* __restrict__ deg, int E) {
  int e = blockIdx.x * blockDim.x + threadIdx.x;
  if (e < E) atomicAdd(&deg[dst[e]], 1.0f);
}

__global__ void gcnt_kernel(const int* __restrict__ batch, float* __restrict__ gcnt, int n) {
  int i = blockIdx.x * blockDim.x + threadIdx.x;
  if (i < n) atomicAdd(&gcnt[batch[i]], 1.0f);
}

// ---------------------------------------------------------------- edge scatter: one wave per edge, 128 channels
__global__ void scatter_edges_kernel(const float* __restrict__ h,
                                     const int* __restrict__ src,
                                     const int* __restrict__ dst,
                                     float* __restrict__ agg, int E) {
  int w    = (blockIdx.x * blockDim.x + threadIdx.x) >> 5;
  int lane = threadIdx.x & 31;
  if (w >= E) return;
  int s = src[w], d = dst[w];
  const float4 v = *reinterpret_cast<const float4*>(h + (size_t)s * kHid + lane * 4);
  float* p = agg + (size_t)d * kHid + lane * 4;
  atomicAdd(p + 0, v.x);
  atomicAdd(p + 1, v.y);
  atomicAdd(p + 2, v.z);
  atomicAdd(p + 3, v.w);
}

// z = h + agg * inv_deg, emitted as bf16 GEMM operand
__global__ void combine_kernel(const float* __restrict__ h,
                               const float* __restrict__ agg,
                               const float* __restrict__ deg,
                               unsigned short* __restrict__ abf, long n) {
  long i = (long)blockIdx.x * blockDim.x + threadIdx.x;
  if (i >= n) return;
  int node  = (int)(i >> 7);
  float inv = 1.0f / fmaxf(deg[node], 1.0f);
  abf[i] = f2bf(h[i] + agg[i] * inv);
}

// ---------------------------------------------------------------- weight transpose + f32->bf16 (Wt[n*K+k] = W[k*N+n])
__global__ void wtrans_kernel(const float* __restrict__ W, unsigned short* __restrict__ Wt,
                              int K, int N) {
  int i = blockIdx.x * blockDim.x + threadIdx.x;
  if (i >= K * N) return;
  int k = i / N, n = i % N;
  Wt[(size_t)n * K + k] = f2bf(W[i]);
}

// ---------------------------------------------------------------- bf16 WMMA GEMM: C[M,N] = A[M,K] @ Bt[N,K]^T + bias
// Whole weight matrix (N*K == 32768 bf16 == 64 KB) staged in LDS once per block
// with an XOR chunk swizzle; A fragments hoisted into VGPRs for all of K; inner
// loop is pure ds_load_b128 -> v_wmma_f32_16x16x32_bf16. One wave owns a 16x64 slab.
template <int K, bool RELU>
__global__ __launch_bounds__(256) void gemm_wmma_kernel(const unsigned short* __restrict__ A_,
                                                        const unsigned short* __restrict__ Bt_,
                                                        const float* __restrict__ bias,
                                                        float* __restrict__ C,
                                                        int M, int N) {
  constexpr int CH = K / 8;                 // 16-byte chunks per B row
  constexpr int NK = K / 32;                // K-steps
  __shared__ __align__(16) unsigned short smem[32768];   // == N*K bf16 for both GEMMs

  // ---- stage Bt (N x K bf16) -> LDS, 16B chunks, XOR-swizzled within each row
  {
    const int chunks = (N * K) / 8;         // 4096
    for (int c = threadIdx.x; c < chunks; c += 256) {
      int row = c / CH, ch = c % CH;
      int dch = ch ^ (row & 7);
      *reinterpret_cast<float4*>(smem + ((size_t)row * CH + dch) * 8) =
          *reinterpret_cast<const float4*>(Bt_ + (size_t)row * K + ch * 8);
    }
  }
  __syncthreads();

  const bf16* A  = reinterpret_cast<const bf16*>(A_);
  const bf16* sm = reinterpret_cast<const bf16*>(smem);

  int wid  = (blockIdx.x * blockDim.x + threadIdx.x) >> 5;
  int lane = threadIdx.x & 31;
  int nGroups = N >> 6;
  int mTiles  = M >> 4;
  int mTile = wid / nGroups;
  int ng    = wid % nGroups;
  if (mTile >= mTiles) return;              // whole-wave uniform exit, after the barrier

  int rlo   = lane & 15;                    // row-in-tile (A) / col-in-tile (B/C)
  int khalf = lane >> 4;
  int arow  = mTile * 16 + rlo;

  // ---- hoist all A fragments for this 16-row strip (16-bit A 16x32 layout:
  //      elems 0..7 -> K = kb+khalf*8.. ; elems 8..15 -> +16)
  bf16x16 afrag[NK];
#pragma unroll
  for (int kk = 0; kk < NK; ++kk) {
    const bf16x8* pa =
        reinterpret_cast<const bf16x8*>(A + (size_t)arow * K + kk * 32 + (khalf << 3));
    afrag[kk] = __builtin_shufflevector(pa[0], pa[2],
        0, 1, 2, 3, 4, 5, 6, 7, 8, 9, 10, 11, 12, 13, 14, 15);
  }

  f32x8 acc[4] = {f32x8{}, f32x8{}, f32x8{}, f32x8{}};

#pragma unroll
  for (int kk = 0; kk < NK; ++kk) {
    int ch0 = kk * 4 + khalf;               // B frag: elems 0..7 K=kb+khalf*8.. ; elems 8..15 -> +16
#pragma unroll
    for (int j = 0; j < 4; ++j) {
      int ncol = ng * 64 + j * 16 + rlo;
      int sw   = ncol & 7;
      bf16x8 lo = *reinterpret_cast<const bf16x8*>(sm + ((size_t)ncol * CH + (ch0 ^ sw)) * 8);
      bf16x8 hi = *reinterpret_cast<const bf16x8*>(sm + ((size_t)ncol * CH + ((ch0 + 2) ^ sw)) * 8);
      bf16x16 bfrag = __builtin_shufflevector(lo, hi,
          0, 1, 2, 3, 4, 5, 6, 7, 8, 9, 10, 11, 12, 13, 14, 15);
      acc[j] = __builtin_amdgcn_wmma_f32_16x16x32_bf16(
          false, afrag[kk], false, bfrag, (short)0, acc[j], false, false);
    }
  }

  // C/D layout: VGPR v -> M = mTile*16 + khalf*8 + v ; lane&15 -> column
  int m0 = mTile * 16 + khalf * 8;
#pragma unroll
  for (int j = 0; j < 4; ++j) {
    int col  = ng * 64 + j * 16 + rlo;
    float bb = bias[col];
#pragma unroll
    for (int v = 0; v < 8; ++v) {
      float val = acc[j][v] + bb;
      if (RELU) val = fmaxf(val, 0.0f);
      C[(size_t)(m0 + v) * N + col] = val;
    }
  }
}

// ---------------------------------------------------------------- BatchNorm (training stats over M rows, C=256)
__global__ void bn_stats_kernel(const float* __restrict__ z, float* __restrict__ bn,
                                int M, int rowsPerBlock) {
  int ch = threadIdx.x;                       // blockDim.x == 256
  int r0 = blockIdx.x * rowsPerBlock;
  int r1 = min(r0 + rowsPerBlock, M);
  float s = 0.0f, s2 = 0.0f;
  for (int r = r0; r < r1; ++r) {
    float v = z[(size_t)r * kHid2 + ch];
    s += v; s2 += v * v;
  }
  atomicAdd(&bn[ch], s);
  atomicAdd(&bn[kHid2 + ch], s2);
}

__global__ void bn_finalize_kernel(float* __restrict__ bn,
                                   const float* __restrict__ gamma,
                                   const float* __restrict__ beta, int M) {
  int ch = threadIdx.x;
  if (ch >= kHid2) return;
  float invM = 1.0f / (float)M;
  float mu   = bn[ch] * invM;
  float var  = bn[kHid2 + ch] * invM - mu * mu;
  float sc   = gamma[ch] * rsqrtf(var + kBnEps);
  bn[2 * kHid2 + ch] = sc;
  bn[3 * kHid2 + ch] = beta[ch] - mu * sc;
}

// normalize + ReLU, emit bf16 operand for the second GEMM
__global__ void bn_apply_kernel(const float* __restrict__ z, const float* __restrict__ bn,
                                unsigned short* __restrict__ abf, long n) {
  long i = (long)blockIdx.x * blockDim.x + threadIdx.x;
  if (i >= n) return;
  int ch  = (int)(i & (kHid2 - 1));
  float v = fmaxf(z[i] * bn[2 * kHid2 + ch] + bn[3 * kHid2 + ch], 0.0f);
  abf[i] = f2bf(v);
}

// ---------------------------------------------------------------- per-layer pooling accumulate into gsum[:, off:off+128]
__global__ void pool_scatter_kernel(const float* __restrict__ h, const int* __restrict__ batch,
                                    float* __restrict__ gsum, int colOff, long n) {
  long i = (long)blockIdx.x * blockDim.x + threadIdx.x;
  if (i >= n) return;
  int node = (int)(i >> 7);
  int c    = (int)(i & (kHid - 1));
  atomicAdd(&gsum[(size_t)batch[node] * kCat + colOff + c], h[i]);
}

// ---------------------------------------------------------------- head: mean-pool finish, FC, log_softmax
__global__ void head_kernel(const float* __restrict__ gsum, const float* __restrict__ gcnt,
                            const float* __restrict__ fcW, const float* __restrict__ fcb,
                            float* __restrict__ out) {
  int g = blockIdx.x * blockDim.x + threadIdx.x;
  if (g >= kGraphs) return;
  float inv = 1.0f / fmaxf(gcnt[g], 1.0f);
  float acc[kClasses];
#pragma unroll
  for (int j = 0; j < kClasses; ++j) acc[j] = fcb[j];
  for (int c = 0; c < kCat; ++c) {
    float p = gsum[(size_t)g * kCat + c] * inv;
#pragma unroll
    for (int j = 0; j < kClasses; ++j) acc[j] += p * fcW[c * kClasses + j];
  }
  float m = acc[0];
#pragma unroll
  for (int j = 1; j < kClasses; ++j) m = fmaxf(m, acc[j]);
  float s = 0.0f;
#pragma unroll
  for (int j = 0; j < kClasses; ++j) s += __expf(acc[j] - m);
  float lse = __logf(s);
#pragma unroll
  for (int j = 0; j < kClasses; ++j) out[g * kClasses + j] = acc[j] - m - lse;
}

// ---------------------------------------------------------------- launch
static inline unsigned int blk(long n, int t) { return (unsigned int)((n + t - 1) / t); }

extern "C" void kernel_launch(void* const* d_in, const int* in_sizes, int n_in,
                              void* d_out, int out_size, void* d_ws, size_t ws_size,
                              hipStream_t stream) {
  (void)in_sizes; (void)n_in; (void)out_size; (void)ws_size;
  const float* x      = (const float*)d_in[0];
  const int*   ei     = (const int*)d_in[1];      // [2, E]
  const int*   batch  = (const int*)d_in[3];
  const float* W1     = (const float*)d_in[4];    // [4,128,256]
  const float* b1     = (const float*)d_in[5];    // [4,256]
  const float* gamma1 = (const float*)d_in[6];
  const float* beta1  = (const float*)d_in[7];
  const float* W2     = (const float*)d_in[8];    // [4,256,128]
  const float* b2     = (const float*)d_in[9];    // [4,128]
  const float* fcW    = (const float*)d_in[10];   // [512,10]
  const float* fcb    = (const float*)d_in[11];
  float* out = (float*)d_out;

  const int* src = ei;
  const int* dst = ei + kEdges;

  // ---- workspace carve-up (all 16B-aligned)
  char* ws = (char*)d_ws;
  size_t off = 0;
  float* gsum = (float*)(ws + off); off += (size_t)kGraphs * kCat * sizeof(float);      // 1 MB
  float* agg  = (float*)(ws + off); off += (size_t)kNodes * kHid * sizeof(float);       // 51.2 MB
  float* hbuf = (float*)(ws + off); off += (size_t)kNodes * kHid * sizeof(float);       // 51.2 MB
  float* zbuf = (float*)(ws + off); off += (size_t)kNodes * kHid2 * sizeof(float);      // 102.4 MB
  unsigned short* abf = (unsigned short*)(ws + off); off += (size_t)kNodes * kHid2 * 2; // 51.2 MB
  unsigned short* w1t = (unsigned short*)(ws + off); off += (size_t)kHid * kHid2 * 2;
  unsigned short* w2t = (unsigned short*)(ws + off); off += (size_t)kHid2 * kHid * 2;
  float* deg  = (float*)(ws + off); off += (size_t)kNodes * sizeof(float);
  float* gcnt = (float*)(ws + off); off += (size_t)kGraphs * sizeof(float);
  float* bn   = (float*)(ws + off); off += 4 * kHid2 * sizeof(float);

  const int T = 256;
  const long nH  = (long)kNodes * kHid;   // 12.8 M
  const long nH2 = (long)kNodes * kHid2;  // 25.6 M

  // ---- graph-level setup
  zerof_kernel<<<blk((long)kGraphs * kCat, T), T, 0, stream>>>(gsum, (long)kGraphs * kCat);
  zerof_kernel<<<blk(kGraphs, T), T, 0, stream>>>(gcnt, kGraphs);
  zerof_kernel<<<blk(kNodes, T), T, 0, stream>>>(deg, kNodes);
  deg_kernel<<<blk(kEdges, T), T, 0, stream>>>(dst, deg, kEdges);
  gcnt_kernel<<<blk(kNodes, T), T, 0, stream>>>(batch, gcnt, kNodes);

  const float* hcur = x;
  for (int layer = 0; layer < kLayers; ++layer) {
    // weights: transpose + bf16
    wtrans_kernel<<<blk(kHid * kHid2, T), T, 0, stream>>>(W1 + (size_t)layer * kHid * kHid2, w1t, kHid, kHid2);
    wtrans_kernel<<<blk(kHid2 * kHid, T), T, 0, stream>>>(W2 + (size_t)layer * kHid2 * kHid, w2t, kHid2, kHid);

    // mean aggregation
    zerof_kernel<<<blk(nH, T), T, 0, stream>>>(agg, nH);
    scatter_edges_kernel<<<blk((long)kEdges * 32, T), T, 0, stream>>>(hcur, src, dst, agg, kEdges);
    combine_kernel<<<blk(nH, T), T, 0, stream>>>(hcur, agg, deg, abf, nH);

    // GEMM1: [N,128] @ [128,256] + b1
    {
      long waves = (long)(kNodes / 16) * (kHid2 / 64);
      gemm_wmma_kernel<kHid, false><<<blk(waves * 32, T), T, 0, stream>>>(
          abf, w1t, b1 + (size_t)layer * kHid2, zbuf, kNodes, kHid2);
    }

    // BatchNorm (training mode) + ReLU -> bf16
    zerof_kernel<<<blk(2 * kHid2, T), T, 0, stream>>>(bn, 2 * kHid2);
    bn_stats_kernel<<<blk(kNodes, 256) /* rowsPerBlock=256 */, kHid2, 0, stream>>>(zbuf, bn, kNodes, 256);
    bn_finalize_kernel<<<1, kHid2, 0, stream>>>(bn, gamma1 + (size_t)layer * kHid2,
                                                beta1 + (size_t)layer * kHid2, kNodes);
    bn_apply_kernel<<<blk(nH2, T), T, 0, stream>>>(zbuf, bn, abf, nH2);

    // GEMM2: [N,256] @ [256,128] + b2, ReLU
    {
      long waves = (long)(kNodes / 16) * (kHid / 64);
      gemm_wmma_kernel<kHid2, true><<<blk(waves * 32, T), T, 0, stream>>>(
          abf, w2t, b2 + (size_t)layer * kHid, hbuf, kNodes, kHid);
    }

    // jumping-knowledge pooling accumulate
    pool_scatter_kernel<<<blk(nH, T), T, 0, stream>>>(hbuf, batch, gsum, layer * kHid, nH);
    hcur = hbuf;
  }

  head_kernel<<<blk(kGraphs, T), T, 0, stream>>>(gsum, gcnt, fcW, fcb, out);
}
